// Model_32246614459085
// MI455X (gfx1250) — compile-verified, matrix operand order checked by reading, not run on hardware
//
#include <hip/hip_runtime.h>
#include <cstdint>
#include <cstddef>

// ---------------- problem constants (from reference) ----------------
#define NN   16384          // nodes
#define BB   512            // graphs
#define NEG  32768          // non-edges
#define PP   16384          // undirected edges
#define EE   32768          // directed edges (2*P)
#define MM   (NN + BB)      // 16896 rows (nodes + cond rows)
#define E0C  (EE + 2*NN)    // 65536 edges before self loops
#define EAUG (E0C + MM)     // 82432 augmented edges

typedef __attribute__((ext_vector_type(16))) __bf16 v16bf;
typedef __attribute__((ext_vector_type(8)))  float  v8f;

union GtFrag { v16bf v; uint4 u[2]; };
union GtPack { uint4 u; __bf16 e[8]; };

// ---------------- small device helpers ----------------
__device__ __forceinline__ float gt_act(float x, int act) {
  if (act == 1) return x > 0.f ? x : 0.01f * x;   // leaky_relu (jax default 0.01)
  if (act == 2) return x > 0.f ? x : 0.f;         // relu
  return x;
}

__device__ __forceinline__ void atomicMaxF(float* addr, float val) {
  unsigned int* ua = (unsigned int*)addr;
  unsigned int old = *ua;
  while (true) {
    float cur = __uint_as_float(old);
    if (cur >= val) break;
    unsigned int assumed = old;
    old = atomicCAS(ua, assumed, __float_as_uint(val));
    if (old == assumed) break;
  }
}

template <bool GUARD>
__device__ __forceinline__ void gt_epi(v8f acc, int row0, int col,
                                       const float* __restrict__ bias, const float* R,
                                       float* C, int ldc, int Ms, int Ns, int act) {
  float bv = 0.f;
  if (bias && (!GUARD || col < Ns)) bv = bias[col];
#pragma unroll
  for (int r = 0; r < 8; ++r) {
    int row = row0 + r;
    if (GUARD && (row >= Ms || col >= Ns)) continue;
    float v = acc[r] + bv;
    if (R) v += R[(size_t)row * ldc + col];
    C[(size_t)row * ldc + col] = gt_act(v, act);
  }
}

// ---------------- bf16 WMMA GEMM: C = act(A@W + bias + R) ----------------
// 128x64 block tile, 8 waves, each wave owns a 32x32 output block (4 WMMAs/K-step).
// A: [Ms,Ks] fp32 (lda); W: [Ks,Ns] fp32 row-major (ldw; pointer pre-offset for col
// slices); bias [Ns] or null; R residual with C's layout or null. fp32->bf16 at LDS
// stage time, fp32 accumulate. GUARD=false: no bounds code at all (all hot shapes).
// GUARD=true: branchless clamped loads (no exec-mask ladders), guarded stores.
template <bool GUARD>
__global__ __launch_bounds__(256) void gt_gemm(
    const float* __restrict__ A, int lda,
    const float* __restrict__ W, int ldw,
    const float* __restrict__ bias,
    const float* R, float* C, int ldc,
    int Ms, int Ns, int Ks, int act)
{
  __shared__ __bf16 As[128][32];   // 8 KB, [m][k]
  __shared__ __bf16 Bt[64][32];    // 4 KB, transposed: [n][k]
  const int tid   = threadIdx.x;
  const int lane  = tid & 31;
  const int wv    = tid >> 5;       // 8 waves
  const int wm    = wv >> 1;        // 0..3  -> 32-row band
  const int wn    = wv & 1;         // 0..1  -> 32-col band
  const int tileM = blockIdx.y * 128;
  const int tileN = blockIdx.x * 64;

  v8f acc00 = {}, acc01 = {}, acc10 = {}, acc11 = {};

  // staging maps
  const int arow = tid >> 1;            // 0..127
  const int acol = (tid & 1) * 16;      // 0 or 16 (16 consecutive k per thread)
  const int bn   = tid & 63;            // 0..63 (column of W within tile)
  const int bk   = (tid >> 6) * 8;      // 0,8,16,24 (8 consecutive k per thread)
  const int kb   = (lane >> 4) * 8;     // K sub-block per half-wave
  const int l15  = lane & 15;

  for (int k0 = 0; k0 < Ks; k0 += 32) {
    // ---- stage A (fp32 -> bf16) ----
    if (!GUARD) {
      const float4* ap = (const float4*)(A + (size_t)(tileM + arow) * lda + k0 + acol);
      float4 f0 = ap[0], f1 = ap[1], f2 = ap[2], f3 = ap[3];
      GtPack p0, p1;
      p0.e[0] = (__bf16)f0.x; p0.e[1] = (__bf16)f0.y; p0.e[2] = (__bf16)f0.z; p0.e[3] = (__bf16)f0.w;
      p0.e[4] = (__bf16)f1.x; p0.e[5] = (__bf16)f1.y; p0.e[6] = (__bf16)f1.z; p0.e[7] = (__bf16)f1.w;
      p1.e[0] = (__bf16)f2.x; p1.e[1] = (__bf16)f2.y; p1.e[2] = (__bf16)f2.z; p1.e[3] = (__bf16)f2.w;
      p1.e[4] = (__bf16)f3.x; p1.e[5] = (__bf16)f3.y; p1.e[6] = (__bf16)f3.z; p1.e[7] = (__bf16)f3.w;
      *(uint4*)&As[arow][acol]     = p0.u;
      *(uint4*)&As[arow][acol + 8] = p1.u;
    } else {
      int gm  = tileM + arow;
      int gmc = gm < Ms ? gm : Ms - 1;          // clamped: always a valid address
      bool mok = (gm < Ms);
#pragma unroll
      for (int j = 0; j < 16; ++j) {
        int gk  = k0 + acol + j;
        int gkc = gk < Ks ? gk : Ks - 1;
        float v = A[(size_t)gmc * lda + gkc];   // unconditional load
        v = (mok && gk < Ks) ? v : 0.f;         // branchless select
        As[arow][acol + j] = (__bf16)v;
      }
    }
    // ---- stage W transposed (fp32 -> bf16) ----
    if (!GUARD) {
      const float* wp = W + (size_t)(k0 + bk) * ldw + tileN + bn;
      GtPack p;
#pragma unroll
      for (int j = 0; j < 8; ++j) p.e[j] = (__bf16)wp[(size_t)j * ldw];
      *(uint4*)&Bt[bn][bk] = p.u;
    } else {
      int gn  = tileN + bn;
      int gnc = gn < Ns ? gn : Ns - 1;
      bool nok = (gn < Ns);
#pragma unroll
      for (int j = 0; j < 8; ++j) {
        int gk  = k0 + bk + j;
        int gkc = gk < Ks ? gk : Ks - 1;
        float v = W[(size_t)gkc * ldw + gnc];
        v = (nok && gk < Ks) ? v : 0.f;
        Bt[bn][bk + j] = (__bf16)v;
      }
    }
    __syncthreads();
    // ---- fragments: contiguous 16B LDS loads ----
    GtFrag a0, a1, b0, b1;
    a0.u[0] = *(const uint4*)&As[wm * 32 + l15][kb];
    a0.u[1] = *(const uint4*)&As[wm * 32 + l15][kb + 16];
    a1.u[0] = *(const uint4*)&As[wm * 32 + 16 + l15][kb];
    a1.u[1] = *(const uint4*)&As[wm * 32 + 16 + l15][kb + 16];
    b0.u[0] = *(const uint4*)&Bt[wn * 32 + l15][kb];
    b0.u[1] = *(const uint4*)&Bt[wn * 32 + l15][kb + 16];
    b1.u[0] = *(const uint4*)&Bt[wn * 32 + 16 + l15][kb];
    b1.u[1] = *(const uint4*)&Bt[wn * 32 + 16 + l15][kb + 16];
    acc00 = __builtin_amdgcn_wmma_f32_16x16x32_bf16(false, a0.v, false, b0.v, (short)0, acc00, false, false);
    acc01 = __builtin_amdgcn_wmma_f32_16x16x32_bf16(false, a0.v, false, b1.v, (short)0, acc01, false, false);
    acc10 = __builtin_amdgcn_wmma_f32_16x16x32_bf16(false, a1.v, false, b0.v, (short)0, acc10, false, false);
    acc11 = __builtin_amdgcn_wmma_f32_16x16x32_bf16(false, a1.v, false, b1.v, (short)0, acc11, false, false);
    __syncthreads();
  }

  const int rb = (lane >> 4) * 8;
  const int r0 = tileM + wm * 32 + rb;
  const int c0 = tileN + wn * 32 + l15;
  gt_epi<GUARD>(acc00, r0,      c0,      bias, R, C, ldc, Ms, Ns, act);
  gt_epi<GUARD>(acc01, r0,      c0 + 16, bias, R, C, ldc, Ms, Ns, act);
  gt_epi<GUARD>(acc10, r0 + 16, c0,      bias, R, C, ldc, Ms, Ns, act);
  gt_epi<GUARD>(acc11, r0 + 16, c0 + 16, bias, R, C, ldc, Ms, Ns, act);
}

// ---------------- elementwise / graph kernels ----------------
__global__ void gt_fill(float* p, float v, size_t n) {
  size_t i = (size_t)blockIdx.x * 256 + threadIdx.x;
  if (i < n) p[i] = v;
}
__global__ void gt_copy(const float* s, float* d, size_t n) {
  size_t i = (size_t)blockIdx.x * 256 + threadIdx.x;
  if (i < n) d[i] = s[i];
}
__global__ void gt_add(float* a, const float* b, size_t n) {
  size_t i = (size_t)blockIdx.x * 256 + threadIdx.x;
  if (i < n) a[i] += b[i];
}
// build src/dst arrays for augmented edge list
__global__ void gt_build(const int* __restrict__ ei, const int* __restrict__ batch,
                         int* srcA, int* dstA) {
  int i = blockIdx.x * 256 + threadIdx.x;
  if (i < EE)            { srcA[i] = ei[i];            dstA[i] = ei[EE + i]; }
  else if (i < EE + NN)  { int j = i - EE;       srcA[i] = j;              dstA[i] = batch[j] + NN; }
  else if (i < E0C)      { int j = i - EE - NN;  srcA[i] = batch[j] + NN;  dstA[i] = j; }
  else if (i < EAUG)     { int j = i - E0C;      srcA[i] = j;              dstA[i] = j; }
}
// e_p rows (E..E+2N): one-hot column 0
__global__ void gt_ep(float* auge) {
  size_t gid = (size_t)blockIdx.x * 256 + threadIdx.x;
  if (gid >= (size_t)2 * NN * 32) return;
  size_t r = gid >> 5; int d4 = (gid & 31) * 4;
  float4 v = make_float4(d4 == 0 ? 1.f : 0.f, 0.f, 0.f, 0.f);
  *(float4*)&auge[((size_t)(EE + r) * 128 + d4)] = v;
}
// loop_attr accumulation (first E0C rows) + degree count
__global__ void gt_loop_acc(const float* __restrict__ auge, const int* __restrict__ dstA,
                            float* ls, float* cnt) {
  size_t gid = (size_t)blockIdx.x * 256 + threadIdx.x;
  if (gid >= (size_t)E0C * 32) return;
  size_t i = gid >> 5; int d4 = (gid & 31) * 4;
  int dn = dstA[i];
  float4 v = *(const float4*)&auge[(size_t)i * 128 + d4];
  float* p = &ls[(size_t)dn * 128 + d4];
  atomicAdd(p + 0, v.x); atomicAdd(p + 1, v.y); atomicAdd(p + 2, v.z); atomicAdd(p + 3, v.w);
  if (d4 == 0) atomicAdd(&cnt[dn], 1.0f);
}
__global__ void gt_loop_fin(const float* __restrict__ ls, const float* __restrict__ cnt,
                            float* auge) {
  size_t gid = (size_t)blockIdx.x * 256 + threadIdx.x;
  if (gid >= (size_t)MM * 32) return;
  size_t r = gid >> 5; int d4 = (gid & 31) * 4;
  float inv = 1.f / fmaxf(cnt[r], 1.0f);
  float4 v = *(const float4*)&ls[(size_t)r * 128 + d4];
  v.x *= inv; v.y *= inv; v.z *= inv; v.w *= inv;
  *(float4*)&auge[(size_t)(E0C + r) * 128 + d4] = v;
}
// msg = relu(o[src] + aug_e) + 1e-7, segment-summed by dst
__global__ void gt_msg(const float* __restrict__ o, const float* __restrict__ auge,
                       const int* __restrict__ srcA, const int* __restrict__ dstA,
                       float* ms) {
  size_t gid = (size_t)blockIdx.x * 256 + threadIdx.x;
  if (gid >= (size_t)EAUG * 32) return;
  size_t i = gid >> 5; int d4 = (gid & 31) * 4;
  float4 a = *(const float4*)&o[(size_t)srcA[i] * 128 + d4];
  float4 b = *(const float4*)&auge[(size_t)i * 128 + d4];
  float* p = &ms[(size_t)dstA[i] * 128 + d4];
  atomicAdd(p + 0, fmaxf(a.x + b.x, 0.f) + 1e-7f);
  atomicAdd(p + 1, fmaxf(a.y + b.y, 0.f) + 1e-7f);
  atomicAdd(p + 2, fmaxf(a.z + b.z, 0.f) + 1e-7f);
  atomicAdd(p + 3, fmaxf(a.w + b.w, 0.f) + 1e-7f);
}
__global__ void gt_concat(const float* __restrict__ o, const float* __restrict__ agg,
                          float* h) {
  size_t gid = (size_t)blockIdx.x * 256 + threadIdx.x;
  if (gid >= (size_t)MM * 64) return;
  size_t r = gid >> 6; int c4 = (gid & 63) * 4;
  float4 v = (c4 < 128) ? *(const float4*)&o[r * 128 + c4]
                        : *(const float4*)&agg[r * 128 + (c4 - 128)];
  *(float4*)&h[r * 256 + c4] = v;
}
// alpha[e] = scale * dot(q[dst], k[src] + eh[e])  (one thread per edge)
__global__ void gt_alpha(const float* __restrict__ qh, const float* __restrict__ kh,
                         const float* __restrict__ eh, const int* __restrict__ srcA,
                         const int* __restrict__ dstA, float* al) {
  int i = blockIdx.x * 256 + threadIdx.x;
  if (i >= EAUG) return;
  const float4* qp = (const float4*)(qh + (size_t)dstA[i] * 128);
  const float4* kp = (const float4*)(kh + (size_t)srcA[i] * 128);
  const float4* ep = (const float4*)(eh + (size_t)i * 128);
  float s = 0.f;
#pragma unroll 4
  for (int d = 0; d < 32; ++d) {
    float4 q = qp[d], k = kp[d], e = ep[d];
    s += q.x * (k.x + e.x) + q.y * (k.y + e.y) + q.z * (k.z + e.z) + q.w * (k.w + e.w);
  }
  al[i] = s * 0.08838834764831845f;   // 1/sqrt(128)
}
__global__ void gt_segmax(const float* __restrict__ al, const int* __restrict__ dstA,
                          float* amax) {
  int i = blockIdx.x * 256 + threadIdx.x;
  if (i >= EAUG) return;
  atomicMaxF(&amax[dstA[i]], al[i]);
}
__global__ void gt_expsum(float* al, const float* __restrict__ amax,
                          const int* __restrict__ dstA, float* den) {
  int i = blockIdx.x * 256 + threadIdx.x;
  if (i >= EAUG) return;
  float a = __expf(al[i] - amax[dstA[i]]);
  al[i] = a;
  atomicAdd(&den[dstA[i]], a);
}
__global__ void gt_tout(const float* __restrict__ al, const float* __restrict__ den,
                        const float* __restrict__ vh, const float* __restrict__ eh,
                        const int* __restrict__ srcA, const int* __restrict__ dstA,
                        float* tout, int hoff) {
  size_t gid = (size_t)blockIdx.x * 256 + threadIdx.x;
  if (gid >= (size_t)EAUG * 32) return;
  size_t i = gid >> 5; int d4 = (gid & 31) * 4;
  int dn = dstA[i];
  float w = al[i] / den[dn];
  float4 v = *(const float4*)&vh[(size_t)srcA[i] * 128 + d4];
  float4 e = *(const float4*)&eh[(size_t)i * 128 + d4];
  float* p = &tout[(size_t)dn * 512 + hoff + d4];
  atomicAdd(p + 0, w * (v.x + e.x));
  atomicAdd(p + 1, w * (v.y + e.y));
  atomicAdd(p + 2, w * (v.z + e.z));
  atomicAdd(p + 3, w * (v.w + e.w));
}
// graph layer-norm over 33 rows x 128 dims per graph (32 nodes + 1 cond row)
__global__ __launch_bounds__(256) void gt_gln(const float* __restrict__ x, float* out,
                                              const float* __restrict__ w,
                                              const float* __restrict__ b) {
  __shared__ float red[256];
  int g = blockIdx.x, tid = threadIdx.x;
  float s = 0.f;
  for (int idx = tid; idx < 33 * 128; idx += 256) {
    int r = idx >> 7, d = idx & 127;
    size_t row = (r < 32) ? (size_t)g * 32 + r : (size_t)NN + g;
    s += x[row * 128 + d];
  }
  red[tid] = s; __syncthreads();
  for (int o = 128; o > 0; o >>= 1) { if (tid < o) red[tid] += red[tid + o]; __syncthreads(); }
  float mean = red[0] * (1.f / 4224.f);
  __syncthreads();
  float v = 0.f;
  for (int idx = tid; idx < 33 * 128; idx += 256) {
    int r = idx >> 7, d = idx & 127;
    size_t row = (r < 32) ? (size_t)g * 32 + r : (size_t)NN + g;
    float xc = x[row * 128 + d] - mean; v += xc * xc;
  }
  red[tid] = v; __syncthreads();
  for (int o = 128; o > 0; o >>= 1) { if (tid < o) red[tid] += red[tid + o]; __syncthreads(); }
  float inv = rsqrtf(red[0] * (1.f / 4224.f) + 1e-5f);
  for (int idx = tid; idx < 33 * 128; idx += 256) {
    int r = idx >> 7, d = idx & 127;
    size_t row = (r < 32) ? (size_t)g * 32 + r : (size_t)NN + g;
    out[row * 128 + d] = (x[row * 128 + d] - mean) * inv * w[d] + b[d];
  }
}
// glob = [gmean | o[N+g] | c[g]]  (B x 384)
__global__ void gt_glob(const float* __restrict__ o, const float* __restrict__ c,
                        float* glob) {
  int g = blockIdx.x, d = threadIdx.x;  // 128 threads
  float s = 0.f;
  for (int r = 0; r < 32; ++r) s += o[(size_t)(g * 32 + r) * 128 + d];
  glob[(size_t)g * 384 + d]       = s * (1.f / 32.f);
  glob[(size_t)g * 384 + 128 + d] = o[(size_t)(NN + g) * 128 + d];
  glob[(size_t)g * 384 + 256 + d] = c[(size_t)g * 128 + d];
}
// out[i] = on[ia[i*sa]] + on[ib[i*sb]]  (n x 128)
__global__ void gt_pair(const float* __restrict__ on, const int* __restrict__ ia, int sa,
                        const int* __restrict__ ib, int sb, float* out, int n) {
  size_t gid = (size_t)blockIdx.x * 256 + threadIdx.x;
  if (gid >= (size_t)n * 32) return;
  size_t i = gid >> 5; int d4 = (gid & 31) * 4;
  float4 a = *(const float4*)&on[(size_t)ia[i * sa] * 128 + d4];
  float4 b = *(const float4*)&on[(size_t)ib[i * sb] * 128 + d4];
  a.x += b.x; a.y += b.y; a.z += b.z; a.w += b.w;
  *(float4*)&out[(size_t)i * 128 + d4] = a;
}

// ---------------- host-side helpers ----------------
static inline dim3 grd(size_t n) { return dim3((unsigned)((n + 255) / 256)); }

static void launch_gemm(hipStream_t st, const float* A, int lda, const float* W, int ldw,
                        const float* bias, const float* R, float* C, int ldc,
                        int Ms, int Ns, int Ks, int act) {
  dim3 grid((unsigned)((Ns + 63) / 64), (unsigned)((Ms + 127) / 128));
  bool full = (Ms % 128 == 0) && (Ns % 64 == 0) && (Ks % 32 == 0);
  if (full) gt_gemm<false><<<grid, 256, 0, st>>>(A, lda, W, ldw, bias, R, C, ldc, Ms, Ns, Ks, act);
  else      gt_gemm<true ><<<grid, 256, 0, st>>>(A, lda, W, ldw, bias, R, C, ldc, Ms, Ns, Ks, act);
}

// 3-layer MLP head: [in,128]->lrelu->[128,128]->lrelu->[128,out]
static void mlp3(hipStream_t st, const float* X, int rows, int in_dim, int ld_in,
                 void* const* d_in, int base, int out_dim,
                 float* t1, float* t2, float* out) {
  const float* W0 = (const float*)d_in[base + 0]; const float* b0 = (const float*)d_in[base + 1];
  const float* W1 = (const float*)d_in[base + 2]; const float* b1 = (const float*)d_in[base + 3];
  const float* W2 = (const float*)d_in[base + 4]; const float* b2 = (const float*)d_in[base + 5];
  launch_gemm(st, X,  ld_in, W0, 128,     b0, nullptr, t1,  128,     rows, 128,     in_dim, 1);
  launch_gemm(st, t1, 128,   W1, 128,     b1, nullptr, t2,  128,     rows, 128,     128,    1);
  launch_gemm(st, t2, 128,   W2, out_dim, b2, nullptr, out, out_dim, rows, out_dim, 128,    0);
}

extern "C" void kernel_launch(void* const* d_in, const int* in_sizes, int n_in,
                              void* d_out, int out_size, void* d_ws, size_t ws_size,
                              hipStream_t stream) {
  (void)in_sizes; (void)out_size; (void)ws_size;
  hipStream_t st = stream;
  const float* xin  = (const float*)d_in[0];           // [N,64]
  const float* ea   = (const float*)d_in[1];           // [E,16]
  const float* cond = (const float*)d_in[2];           // [B,32]
  const int*   ei   = (const int*)d_in[n_in - 3];      // edge_index [2,E]
  const int*   nei  = (const int*)d_in[n_in - 2];      // non_edge_index [2,NE]
  const int*   bat  = (const int*)d_in[n_in - 1];      // batch [N]
  float* dout = (float*)d_out;

  // ---- workspace carve-up (floats) ----
  float* F = (float*)d_ws;
  size_t off = 0;
  auto alloc = [&](size_t n) { size_t r = off; off += n; return r; };
  size_t o_o   = alloc((size_t)MM * 128);
  size_t o_aug = alloc((size_t)EAUG * 128);
  size_t o_ms  = alloc((size_t)MM * 128);     // loopsum / msgsum
  size_t o_cnt = alloc(MM);
  size_t o_amx = alloc(MM);
  size_t o_den = alloc(MM + 2);               // 2-float pad: later offsets are 8B-, not 16B-aligned
  size_t o_h   = alloc((size_t)MM * 256);
  size_t o_g1  = alloc((size_t)MM * 256);
  size_t o_g2  = alloc((size_t)MM * 256);
  size_t o_agg = alloc((size_t)MM * 128);     // reused as glob [B,384] after layers
  size_t o_qh  = alloc((size_t)MM * 128);
  size_t o_kh  = alloc((size_t)MM * 128);
  size_t o_vh  = alloc((size_t)MM * 128);
  size_t o_eh  = alloc((size_t)EAUG * 128);   // reused as pe/se after layers
  size_t o_al  = alloc(EAUG);
  size_t o_to  = alloc((size_t)MM * 512);     // tout; also head tmp1 / encoder tmp1
  size_t o_ff  = alloc((size_t)MM * 512);     // ff hidden; also head tmp2 / encoder tmp2
  size_t o_ln  = alloc((size_t)MM * 128);
  size_t o_c   = alloc((size_t)BB * 128);
  int* srcA = (int*)(F + off); off += EAUG;
  int* dstA = (int*)(F + off); off += EAUG;

  float* o    = F + o_o;
  float* auge = F + o_aug;
  float* ms   = F + o_ms;
  float* cntb = F + o_cnt;
  float* amax = F + o_amx;
  float* den  = F + o_den;
  float* hbuf = F + o_h;
  float* g1b  = F + o_g1;
  float* g2b  = F + o_g2;
  float* agg  = F + o_agg;
  float* qh   = F + o_qh;
  float* kh   = F + o_kh;
  float* vh   = F + o_vh;
  float* ehb  = F + o_eh;
  float* alb  = F + o_al;
  float* tout = F + o_to;
  float* ffh  = F + o_ff;
  float* lnin = F + o_ln;
  float* cbuf = F + o_c;
  float* t1   = F + o_to;   // pre/post-layer scratch reuse
  float* t2   = F + o_ff;

  // ---- graph topology ----
  gt_build<<<grd(EAUG), 256, 0, st>>>(ei, bat, srcA, dstA);

  // ---- encoders (params base indices per JAX sorted-dict flatten) ----
  mlp3(st, xin,  NN, 64, 64, d_in, 201, 128, t1, t2, o);      // x2h -> o[0:N]
  mlp3(st, ea,   EE, 16, 16, d_in,   9, 128, t1, t2, auge);   // e2h -> aug_e[0:E]
  mlp3(st, cond, BB, 32, 32, d_in,   3, 128, t1, t2, cbuf);   // c2h -> c
  gt_copy<<<grd((size_t)BB * 128), 256, 0, st>>>(cbuf, o + (size_t)NN * 128, (size_t)BB * 128);

  // ---- e_p rows + loop_attr (self-loop edge features) ----
  gt_ep<<<grd((size_t)2 * NN * 32), 256, 0, st>>>(auge);
  gt_fill<<<grd((size_t)MM * 128), 256, 0, st>>>(ms, 0.f, (size_t)MM * 128);
  gt_fill<<<grd(MM), 256, 0, st>>>(cntb, 0.f, MM);
  gt_loop_acc<<<grd((size_t)E0C * 32), 256, 0, st>>>(auge, dstA, ms, cntb);
  gt_loop_fin<<<grd((size_t)MM * 32), 256, 0, st>>>(ms, cntb, auge);

  // ---- transformer layers ----
  for (int l = 0; l < 6; ++l) {
    int lb = 51 + 25 * l;   // sorted keys: e,ff,gen_mlp,k,lin,n1_b,n1_w,n2_b,n2_w,q,skip,v
    const float* eW   = (const float*)d_in[lb + 0];
    const float* ffW0 = (const float*)d_in[lb + 1];  const float* ffb0 = (const float*)d_in[lb + 2];
    const float* ffW1 = (const float*)d_in[lb + 3];  const float* ffb1 = (const float*)d_in[lb + 4];
    const float* gW0  = (const float*)d_in[lb + 5];  const float* gb0  = (const float*)d_in[lb + 6];
    const float* gW1  = (const float*)d_in[lb + 7];  const float* gb1  = (const float*)d_in[lb + 8];
    const float* gW2  = (const float*)d_in[lb + 9];  const float* gb2  = (const float*)d_in[lb + 10];
    const float* kW   = (const float*)d_in[lb + 11]; const float* kbv  = (const float*)d_in[lb + 12];
    const float* linW = (const float*)d_in[lb + 13]; const float* linb = (const float*)d_in[lb + 14];
    const float* n1b  = (const float*)d_in[lb + 15]; const float* n1w  = (const float*)d_in[lb + 16];
    const float* n2b  = (const float*)d_in[lb + 17]; const float* n2w  = (const float*)d_in[lb + 18];
    const float* qW   = (const float*)d_in[lb + 19]; const float* qb   = (const float*)d_in[lb + 20];
    const float* skW  = (const float*)d_in[lb + 21]; const float* skb  = (const float*)d_in[lb + 22];
    const float* vW   = (const float*)d_in[lb + 23]; const float* vb   = (const float*)d_in[lb + 24];

    // message pass + gen_mlp
    gt_fill<<<grd((size_t)MM * 128), 256, 0, st>>>(ms, 0.f, (size_t)MM * 128);
    gt_msg<<<grd((size_t)EAUG * 32), 256, 0, st>>>(o, auge, srcA, dstA, ms);
    gt_add<<<grd((size_t)MM * 128), 256, 0, st>>>(ms, o, (size_t)MM * 128);
    launch_gemm(st, ms,  128, gW0, 256, gb0, nullptr, g1b, 256, MM, 256, 128, 2);
    launch_gemm(st, g1b, 256, gW1, 256, gb1, nullptr, g2b, 256, MM, 256, 256, 2);
    launch_gemm(st, g2b, 256, gW2, 128, gb2, nullptr, agg, 128, MM, 128, 256, 0);
    gt_concat<<<grd((size_t)MM * 64), 256, 0, st>>>(o, agg, hbuf);

    // multi-head edge-augmented attention
    gt_fill<<<grd((size_t)MM * 512), 256, 0, st>>>(tout, 0.f, (size_t)MM * 512);
    for (int hd = 0; hd < 4; ++hd) {
      int co = hd * 128;
      launch_gemm(st, hbuf, 256, qW + co, 512, qb + co,  nullptr, qh,  128, MM,   128, 256, 0);
      launch_gemm(st, hbuf, 256, kW + co, 512, kbv + co, nullptr, kh,  128, MM,   128, 256, 0);
      launch_gemm(st, hbuf, 256, vW + co, 512, vb + co,  nullptr, vh,  128, MM,   128, 256, 0);
      launch_gemm(st, auge, 128, eW + co, 512, nullptr,  nullptr, ehb, 128, EAUG, 128, 128, 0);
      gt_alpha<<<grd(EAUG), 256, 0, st>>>(qh, kh, ehb, srcA, dstA, alb);
      gt_fill<<<grd(MM), 256, 0, st>>>(amax, -1e30f, MM);
      gt_fill<<<grd(MM), 256, 0, st>>>(den, 0.f, MM);
      gt_segmax<<<grd(EAUG), 256, 0, st>>>(alb, dstA, amax);
      gt_expsum<<<grd(EAUG), 256, 0, st>>>(alb, amax, dstA, den);
      gt_tout<<<grd((size_t)EAUG * 32), 256, 0, st>>>(alb, den, vh, ehb, srcA, dstA, tout, co);
    }
    // tout += skip(h);  o = GLN(o + lin(tout));  o = GLN(o + FF(o))
    launch_gemm(st, hbuf, 256, skW,  512, skb,  tout, tout, 512, MM, 512, 256, 0);
    launch_gemm(st, tout, 512, linW, 128, linb, o,    lnin, 128, MM, 128, 512, 0);
    gt_gln<<<BB, 256, 0, st>>>(lnin, o, n1w, n1b);
    launch_gemm(st, o,   128, ffW0, 512, ffb0, nullptr, ffh,  512, MM, 512, 128, 1);
    launch_gemm(st, ffh, 512, ffW1, 128, ffb1, o,       lnin, 128, MM, 128, 512, 0);
    gt_gln<<<BB, 256, 0, st>>>(lnin, o, n2w, n2b);
  }

  // ---- output heads ----
  float* glob = agg;                                  // B x 384 (reuse)
  gt_glob<<<BB, 128, 0, st>>>(o, cbuf, glob);
  const size_t O1 = 512;                              // add_node
  const size_t O2 = O1 + (size_t)NN * 72;             // set_node_attr
  const size_t O3 = O2 + (size_t)NN * 32;             // add_edge
  const size_t O4 = O3 + (size_t)NEG * 1;             // set_edge_attr
  const size_t O5 = O4 + (size_t)PP * 16;             // reward

  mlp3(st, glob, BB, 384, 384, d_in, 45, 1,  t1, t2, dout);        // emb2stop
  mlp3(st, o,    NN, 128, 128, d_in, 21, 72, t1, t2, dout + O1);   // emb2add_node
  mlp3(st, o,    NN, 128, 128, d_in, 39, 32, t1, t2, dout + O2);   // emb2set_node_attr
  float* pe = ehb;                                    // NE x 128 (reuse)
  gt_pair<<<grd((size_t)NEG * 32), 256, 0, st>>>(o, nei, 1, nei + NEG, 1, pe, NEG);
  mlp3(st, pe, NEG, 128, 128, d_in, 15, 1, t1, t2, dout + O3);     // emb2add_edge
  float* se = ehb + (size_t)NEG * 128;                // P x 128 (reuse)
  gt_pair<<<grd((size_t)PP * 32), 256, 0, st>>>(o, ei, 2, ei + EE, 2, se, PP);
  mlp3(st, se, PP, 128, 128, d_in, 33, 16, t1, t2, dout + O4);     // emb2set_edge_attr
  mlp3(st, glob, BB, 384, 384, d_in, 27, 1, t1, t2, dout + O5);    // emb2reward
}
